// WavefunctionEmbedding_16630113370598
// MI455X (gfx1250) — matrix operand order repaired and checked by reading, not compiled, not run on hardware
//
#include <hip/hip_runtime.h>

// ---------------------------------------------------------------------------
// Problem constants (from the reference)
// ---------------------------------------------------------------------------
#define BB      4
#define NN      2048
#define DMODEL  512
#define HALF    256
#define DHID    1024
#define LN_EPS  1e-5f
#define TWO_PI  6.283185307179586f

#define USE_ASYNC_LDS 1

typedef __attribute__((ext_vector_type(16))) __bf16 v16bf;
typedef __attribute__((ext_vector_type(8)))  float  v8f;

__device__ __forceinline__ unsigned short f2bf(float f) {
    unsigned int u = __float_as_uint(f);
    unsigned int r = (u + 0x7FFFu + ((u >> 16) & 1u)) >> 16;   // RNE
    return (unsigned short)r;
}

__device__ __forceinline__ float fast_tanh(float x) {
#if __has_builtin(__builtin_amdgcn_tanhf)
    return __builtin_amdgcn_tanhf(x);            // v_tanh_f32 (TRANS)
#else
    float e = __expf(-2.0f * fabsf(x));          // v_exp_f32
    float t = (1.0f - e) * __builtin_amdgcn_rcpf(1.0f + e);
    return copysignf(t, x);
#endif
}

// Async global->LDS copy of 16B (CDNA5 async-LDS pipeline, ISA 15.18.3)
__device__ __forceinline__ void copy16_to_lds(void* lds_dst, const void* gsrc) {
#if USE_ASYNC_LDS
    unsigned lds = (unsigned)(unsigned long long)lds_dst;   // LDS offset = addr[31:0]
    asm volatile("global_load_async_to_lds_b128 %0, %1, off"
                 :: "v"(lds), "v"(gsrc) : "memory");
#else
    *(uint4*)lds_dst = *(const uint4*)gsrc;
#endif
}
__device__ __forceinline__ void async_wait() {
#if USE_ASYNC_LDS
    asm volatile("s_wait_asynccnt 0x0" ::: "memory");
#endif
}

// ---------------------------------------------------------------------------
// Kernel 0: pack fp32 weights -> bf16 in WMMA-fragment-major order:
//   wfrag[((ks*NT + nt)*32 + lane)*8 + v] = { w[k][n], w[k+1][n] }  (lo,hi bf16)
//   with k = ks*32 + 2v + (lane>>4)*16,  n = nt*16 + (lane&15)
// so a lane's whole B fragment for (ks, ntile) is 32 contiguous bytes.
// ---------------------------------------------------------------------------
__global__ __launch_bounds__(256) void pack_w_kernel(
    const float* __restrict__ w, unsigned int* __restrict__ wfrag,
    int N, int NT)
{
    const int i    = blockIdx.x * 256 + threadIdx.x;   // one u32 per thread
    const int v    = i & 7;
    const int lane = (i >> 3) & 31;
    const int nt   = (i >> 8) % NT;
    const int ks   = (i >> 8) / NT;
    const int k    = ks * 32 + 2 * v + (lane >> 4) * 16;
    const int n    = nt * 16 + (lane & 15);
    const unsigned int lo = f2bf(w[(size_t)k * N + n]);
    const unsigned int hi = f2bf(w[(size_t)(k + 1) * N + n]);
    wfrag[i] = lo | (hi << 16);
}

// ---------------------------------------------------------------------------
// Kernel 1: wavefunction embedding.
// grid (N/32, B), block 256.  thread = (query q = tid&31, chunk c = tid>>5)
// Phase computed in *revolutions* (dist/lambda) so native v_sin/v_cos apply
// their implicit 2*pi directly: sin(2*pi*dist/lambda) == sin(k*dist).
// ---------------------------------------------------------------------------
__global__ __launch_bounds__(256) void wf_embed_kernel(
    const float* __restrict__ coords,        // [B,N,3]
    const unsigned char* __restrict__ pmask, // [B,N] 1 = padded
    const float* __restrict__ wl,            // [HALF]
    float* __restrict__ wf)                  // [B,N,DMODEL]
{
    __shared__ float sx[256], sy[256], sz[256], sm[256];

    const int tid = threadIdx.x;
    const int b  = blockIdx.y;
    const int q  = tid & 31;
    const int c  = tid >> 5;                  // wavenumber chunk (0..7)
    const int i  = blockIdx.x * 32 + q;       // query index

    const float qx = coords[(b * NN + i) * 3 + 0];
    const float qy = coords[(b * NN + i) * 3 + 1];
    const float qz = coords[(b * NN + i) * 3 + 2];

    float kv[32];                              // revolutions per unit distance
#pragma unroll
    for (int kk = 0; kk < 32; ++kk)
        kv[kk] = 1.0f / wl[c * 32 + kk];

    float re[32], im[32];
#pragma unroll
    for (int kk = 0; kk < 32; ++kk) { re[kk] = 0.f; im[kk] = 0.f; }

    for (int j0 = 0; j0 < NN; j0 += 256) {
        __syncthreads();
        const int j = j0 + tid;
        sx[tid] = coords[(b * NN + j) * 3 + 0];
        sy[tid] = coords[(b * NN + j) * 3 + 1];
        sz[tid] = coords[(b * NN + j) * 3 + 2];
        sm[tid] = pmask[b * NN + j] ? 0.f : 1.f;
        __syncthreads();

        for (int jj = 0; jj < 256; ++jj) {
            const float dx = qx - sx[jj];
            const float dy = qy - sy[jj];
            const float dz = qz - sz[jj];
            const float d2 = dx * dx + dy * dy + dz * dz;
            const float rinv = (d2 > 0.f) ? __frsqrt_rn(d2) : 0.f;
            const float mag  = rinv * sm[jj];   // 0 for self / padded source
            const float dist = d2 * rinv;
#pragma unroll
            for (int kk = 0; kk < 32; ++kk) {
                const float ph = kv[kk] * dist;  // revolutions
#if __has_builtin(__builtin_amdgcn_sinf) && __has_builtin(__builtin_amdgcn_cosf)
                const float s  = __builtin_amdgcn_sinf(ph);
                const float co = __builtin_amdgcn_cosf(ph);
#else
                float s, co;
                __sincosf(TWO_PI * ph, &s, &co);
#endif
                re[kk] = fmaf(co, mag, re[kk]);
                im[kk] = fmaf(s,  mag, im[kk]);
            }
        }
    }

    const float qv = pmask[b * NN + i] ? 0.f : 1.f;   // zero padded queries
    const size_t base = (size_t)(b * NN + i) * DMODEL;
#pragma unroll
    for (int kk = 0; kk < 32; ++kk) {
        const int k = c * 32 + kk;
        wf[base + 2 * k + 0] = re[kk] * qv;
        wf[base + 2 * k + 1] = im[kk] * qv;
    }
}

// ---------------------------------------------------------------------------
// Kernel 2: LayerNorm1.  One 32-lane wave per row; 8 rows per block.
// ---------------------------------------------------------------------------
__global__ __launch_bounds__(256) void ln1_kernel(
    const float* __restrict__ wf, const float* __restrict__ g,
    const float* __restrict__ be, float* __restrict__ h,
    unsigned short* __restrict__ hb)
{
    const int tid  = threadIdx.x;
    const int lane = tid & 31;
    const int row  = blockIdx.x * 8 + (tid >> 5);

    float x[16];
    float s = 0.f, s2 = 0.f;
#pragma unroll
    for (int qd = 0; qd < 4; ++qd) {
        float4 v = *(const float4*)&wf[(size_t)row * DMODEL + lane * 16 + 4 * qd];
        x[4 * qd + 0] = v.x; x[4 * qd + 1] = v.y;
        x[4 * qd + 2] = v.z; x[4 * qd + 3] = v.w;
    }
#pragma unroll
    for (int e = 0; e < 16; ++e) { s += x[e]; s2 += x[e] * x[e]; }
#pragma unroll
    for (int m = 1; m < 32; m <<= 1) {
        s  += __shfl_xor(s,  m, 32);
        s2 += __shfl_xor(s2, m, 32);
    }
    const float mu   = s * (1.f / DMODEL);
    const float var  = s2 * (1.f / DMODEL) - mu * mu;
    const float rstd = __frsqrt_rn(var + LN_EPS);

#pragma unroll
    for (int e = 0; e < 16; ++e) {
        const int col = lane * 16 + e;
        const float y = g[col] * (x[e] - mu) * rstd + be[col];
        h[(size_t)row * DMODEL + col]  = y;
        hb[(size_t)row * DMODEL + col] = f2bf(y);
    }
}

// ---------------------------------------------------------------------------
// WMMA fragment conventions (A/B in this code's contiguous k order, not AMD's table order):
//   A (16x32 bf16): VGPR r, lane l -> m = l&15, K = 2r + (l>>4)*16
//     => lane reads As[m][kA16 .. kA16+7] and As[m][kA16+8 .. +15] (2x b128)
//   B (32x16 bf16): VGPR v, lane l -> n = l&15, K = 2v + (l>>4)*16 (pair-packed)
//     => fragment pre-packed contiguously in global (pack_w_kernel), 2x b128
//   C/D (16x16 f32): VGPR r, lane l -> m = r + (l>>4)*8, n = l&15
// ---------------------------------------------------------------------------
typedef union { uint4 q[2]; unsigned int u[8]; v16bf v; } frag_t;

// Kernel 3: GEMM1  act = gelu(h_bf16[8192,512] @ w1[512,1024] + b1)
// Block tile 64x128 (8 waves, wave tile 16x64).
// A tile: async global->LDS DMA, double-buffered so the copy for kstep+1
// overlaps this kstep's WMMAs.  B: fragment-major global, register
// double-buffered.
__global__ __launch_bounds__(256) void gemm1_kernel(
    const unsigned short* __restrict__ hb,   // [8192,512] bf16
    const unsigned int*  __restrict__ w1f,   // frag-major [16][64][32][8]
    const float* __restrict__ b1,            // [1024]
    unsigned short* __restrict__ actb)       // [8192,1024] bf16
{
    __shared__ unsigned short As[2][64][32]; // 2 x 4 KB, double-buffered

    const int tid  = threadIdx.x;
    const int wave = tid >> 5;
    const int lane = tid & 31;
    const int m0   = blockIdx.y * 64;
    const int nt0  = blockIdx.x * 8;         // first col-tile (8 tiles = 128 cols)
    const int rw   = (wave & 3) * 16;        // wave row offset inside block
    const int ct   = (wave >> 2) * 4;        // wave's first col-tile within block

    v8f acc[4] = {};
    frag_t bbuf[2][4];

    auto bload = [&](int ks, frag_t* dst) {
#pragma unroll
        for (int t = 0; t < 4; ++t) {
            const uint4* p = (const uint4*)(w1f +
                (((size_t)ks * 64 + nt0 + ct + t) * 32 + lane) * 8);
            dst[t].q[0] = p[0];
            dst[t].q[1] = p[1];
        }
    };
    bload(0, bbuf[0]);

    const int mA   = rw + (lane & 15);
    const int kA16 = (lane >> 4) * 16;       // in u16 units
    const int rA   = tid >> 2;               // A staging: row per thread
    const int cA   = (tid & 3) * 8;          // 8 bf16 = 16B per thread

    // prologue: start DMA of A tile for kstep 0
    copy16_to_lds(&As[0][rA][cA], &hb[(size_t)(m0 + rA) * DMODEL + cA]);

#pragma unroll
    for (int ks = 0; ks < 16; ++ks) {
        async_wait();                        // A tile for this kstep has landed
        __syncthreads();                     // publish; prior readers finished

        frag_t af;
        af.q[0] = *(const uint4*)&As[ks & 1][mA][kA16];
        af.q[1] = *(const uint4*)&As[ks & 1][mA][kA16 + 8];

        if (ks < 15) {
            // start DMA for next kstep into the other buffer (last read ks-1)
            copy16_to_lds(&As[(ks + 1) & 1][rA][cA],
                          &hb[(size_t)(m0 + rA) * DMODEL + (ks + 1) * 32 + cA]);
            bload(ks + 1, bbuf[(ks + 1) & 1]);
        }

#pragma unroll
        for (int t = 0; t < 4; ++t)
            acc[t] = __builtin_amdgcn_wmma_f32_16x16x32_bf16(
                false, af.v, false, bbuf[ks & 1][t].v, (short)0, acc[t], false, false);
    }

    const int mOut = m0 + rw + (lane >> 4) * 8;
#pragma unroll
    for (int t = 0; t < 4; ++t) {
        const int n = (nt0 + ct + t) * 16 + (lane & 15);
        const float bias = b1[n];
#pragma unroll
        for (int r = 0; r < 8; ++r) {
            float x = acc[t][r] + bias;
            float inner = 0.7978845608028654f * (x + 0.044715f * x * x * x);
            float gl = 0.5f * x * (1.0f + fast_tanh(inner));  // tanh-GELU
            actb[(size_t)(mOut + r) * DHID + n] = f2bf(gl);
        }
    }
}

// Kernel 4: GEMM2 + bias + residual + LayerNorm2.
// Block computes 16 rows x full 512 cols (8 waves x 4 col-tiles).
__global__ __launch_bounds__(256) void gemm2_ln_kernel(
    const unsigned short* __restrict__ actb, // [8192,1024] bf16
    const unsigned int*  __restrict__ w2f,   // frag-major [32][32][32][8]
    const float* __restrict__ b2,            // [512]
    const float* __restrict__ hres,          // [8192,512] f32 (residual)
    const float* __restrict__ g2, const float* __restrict__ be2,
    float* __restrict__ out)                 // [8192,512]
{
    __shared__ unsigned short As[2][16][32]; // 2 x 1 KB, double-buffered
    __shared__ float lns[16 * 512];          // 32 KB LN staging

    const int tid  = threadIdx.x;
    const int wave = tid >> 5;
    const int lane = tid & 31;
    const int m0   = blockIdx.x * 16;
    const int ct   = wave * 4;               // wave's first col-tile (of 32)

    v8f acc[4] = {};
    frag_t bbuf[2][4];

    auto bload = [&](int ks, frag_t* dst) {
#pragma unroll
        for (int t = 0; t < 4; ++t) {
            const uint4* p = (const uint4*)(w2f +
                (((size_t)ks * 32 + ct + t) * 32 + lane) * 8);
            dst[t].q[0] = p[0];
            dst[t].q[1] = p[1];
        }
    };
    bload(0, bbuf[0]);

    const int mA   = lane & 15;
    const int kA16 = (lane >> 4) * 16;
    const int rA   = tid >> 2;               // rows 0..15 for tid<64
    const int cA   = (tid & 3) * 8;

    if (tid < 64)
        copy16_to_lds(&As[0][rA][cA], &actb[(size_t)(m0 + rA) * DHID + cA]);

#pragma unroll
    for (int ks = 0; ks < 32; ++ks) {
        async_wait();
        __syncthreads();

        frag_t af;
        af.q[0] = *(const uint4*)&As[ks & 1][mA][kA16];
        af.q[1] = *(const uint4*)&As[ks & 1][mA][kA16 + 8];

        if (ks < 31) {
            if (tid < 64)
                copy16_to_lds(&As[(ks + 1) & 1][rA][cA],
                              &actb[(size_t)(m0 + rA) * DHID + (ks + 1) * 32 + cA]);
            bload(ks + 1, bbuf[(ks + 1) & 1]);
        }

#pragma unroll
        for (int t = 0; t < 4; ++t)
            acc[t] = __builtin_amdgcn_wmma_f32_16x16x32_bf16(
                false, af.v, false, bbuf[ks & 1][t].v, (short)0, acc[t], false, false);
    }

    // bias + residual -> LN staging
    __syncthreads();
    const int mloc = (lane >> 4) * 8;
#pragma unroll
    for (int t = 0; t < 4; ++t) {
        const int n = (ct + t) * 16 + (lane & 15);
        const float bias = b2[n];
#pragma unroll
        for (int r = 0; r < 8; ++r) {
            const int m = mloc + r;
            const float x = acc[t][r] + bias + hres[(size_t)(m0 + m) * DMODEL + n];
            lns[m * 512 + n] = x;
        }
    }
    __syncthreads();

    // LayerNorm per row: 16 threads per row, 32 cols each
    const int row = tid >> 4;
    const int sub = tid & 15;
    float s = 0.f, s2 = 0.f;
#pragma unroll
    for (int cc = 0; cc < 32; ++cc) {
        const float x = lns[row * 512 + sub * 32 + cc];
        s += x; s2 += x * x;
    }
#pragma unroll
    for (int m = 1; m < 16; m <<= 1) {       // stays within 16-lane half-wave
        s  += __shfl_xor(s,  m, 32);
        s2 += __shfl_xor(s2, m, 32);
    }
    const float mu   = s * (1.f / DMODEL);
    const float var  = s2 * (1.f / DMODEL) - mu * mu;
    const float rstd = __frsqrt_rn(var + LN_EPS);
#pragma unroll
    for (int cc = 0; cc < 32; ++cc) {
        const int n = sub * 32 + cc;
        const float x = lns[row * 512 + n];
        out[(size_t)(m0 + row) * DMODEL + n] = g2[n] * (x - mu) * rstd + be2[n];
    }
}

// ---------------------------------------------------------------------------
// Host-side launcher
// ---------------------------------------------------------------------------
extern "C" void kernel_launch(void* const* d_in, const int* in_sizes, int n_in,
                              void* d_out, int out_size, void* d_ws, size_t ws_size,
                              hipStream_t stream)
{
    (void)in_sizes; (void)n_in; (void)out_size; (void)ws_size;

    const float*         coords = (const float*)d_in[0];
    const unsigned char* pmask  = (const unsigned char*)d_in[1];
    const float*         wl     = (const float*)d_in[2];
    const float*         w1     = (const float*)d_in[3];
    const float*         b1     = (const float*)d_in[4];
    const float*         w2     = (const float*)d_in[5];
    const float*         b2     = (const float*)d_in[6];
    const float*         ln1g   = (const float*)d_in[7];
    const float*         ln1b   = (const float*)d_in[8];
    const float*         ln2g   = (const float*)d_in[9];
    const float*         ln2b   = (const float*)d_in[10];
    float*               out    = (float*)d_out;

    const size_t tokens = (size_t)BB * NN;            // 8192
    const size_t wu32   = (size_t)DMODEL * DHID / 2;  // 262144 u32 per weight

    float*          wf   = (float*)d_ws;              // tokens*512 f32
    float*          hres = wf + tokens * DMODEL;      // tokens*512 f32
    unsigned short* hb   = (unsigned short*)(hres + tokens * DMODEL);
    unsigned int*   w1f  = (unsigned int*)(hb + tokens * DMODEL);
    unsigned int*   w2f  = w1f + wu32;
    unsigned short* actb = (unsigned short*)(w2f + wu32);  // tokens*1024 bf16

    // 0) weights -> bf16, fragment-major
    pack_w_kernel<<<(unsigned)(wu32 / 256), 256, 0, stream>>>(w1, w1f, DHID,   DHID / 16);
    pack_w_kernel<<<(unsigned)(wu32 / 256), 256, 0, stream>>>(w2, w2f, DMODEL, DMODEL / 16);

    // 1) wavefunction embedding
    wf_embed_kernel<<<dim3(NN / 32, BB), 256, 0, stream>>>(coords, pmask, wl, wf);

    // 2) LayerNorm1 -> hres (f32) + hb (bf16)
    ln1_kernel<<<(unsigned)(tokens / 8), 256, 0, stream>>>(wf, ln1g, ln1b, hres, hb);

    // 3) GEMM1 + bias + gelu -> actb
    gemm1_kernel<<<dim3(DHID / 128, (unsigned)(tokens / 64)), 256, 0, stream>>>(hb, w1f, b1, actb);

    // 4) GEMM2 + bias + residual + LayerNorm2 -> out
    gemm2_ln_kernel<<<(unsigned)(tokens / 16), 256, 0, stream>>>(actb, w2f, b2, hres, ln2g, ln2b, out);
}
